// ResidualVectorQuantizer_3178275799664
// MI455X (gfx1250) — matrix operand order repaired and checked by reading, not compiled, not run on hardware
//
#include <hip/hip_runtime.h>
#include <hip/hip_bf16.h>

typedef __attribute__((ext_vector_type(16))) __bf16 v16bf;
typedef __attribute__((ext_vector_type(8)))  float  v8f;

#define N_VEC   65536
#define K_CB    1024
#define DIM     256
#define LEVELS  4
#define LDS_PITCH 260          // 256 floats + 4 pad -> conflict-free banks

// ---- workspace layout (bytes) ----
#define WS_BPACK   0u          // 4*64*8*32*16 * 2B = 2,097,152 (bf16 B-operand swizzled)
#define WS_WNORM   2097152u    // 4096 * 4B
#define WS_COUNTS  2113536u    // 4096 * 4B
#define WS_IDX     2129920u    // 4*65536 * 4B = 1,048,576
#define WS_SUMSQ   3178496u    // 4 * 4B (padded)
#define WS_PAIRS   3178560u    // 4 * 4B

// ---- output layout (floats) ----
#define OUT_SCAL   16777216    // quantized = [0, 16777216)
#define OUT_IDX    16777219    // 4*65536 index values as float

// ---------------- init: zero accumulators ----------------
__global__ void init_k(unsigned* __restrict__ counts, float* __restrict__ sumsq,
                       float* __restrict__ pairs) {
  int t = threadIdx.x;
  for (int k = t; k < LEVELS * K_CB; k += 256) counts[k] = 0u;
  if (t < LEVELS) { sumsq[t] = 0.f; pairs[t] = 0.f; }
}

// ---------------- pack codebooks into WMMA B-operand bf16 layout ----------------
// B tile 32x16, k order chosen to match A: lanes 0-15 cols n, K=kb..kb+15 ; lanes 16-31 cols n, K=kb+16..kb+31
__global__ void pack_k(const float* __restrict__ cb, __bf16* __restrict__ bp) {
  int t    = blockIdx.x * blockDim.x + threadIdx.x;   // 65536 threads
  int lane = t & 31;
  int kc   = (t >> 5) & 7;
  int tile = (t >> 8) & 63;
  int c    = t >> 14;
  int row  = tile * 16 + (lane & 15);
  int kb   = kc * 32 + ((lane & 16) ? 16 : 0);
  const float* src = cb + ((size_t)(c * K_CB + row)) * DIM + kb;
  __bf16* dst = bp + (size_t)t * 16;   // t == (((c*64+tile)*8)+kc)*32 + lane
#pragma unroll
  for (int h = 0; h < 16; ++h) dst[h] = (__bf16)src[h];
}

// ---------------- per-codeword squared norms ----------------
__global__ void wnorm_k(const float* __restrict__ cb, float* __restrict__ wn) {
  int t = blockIdx.x * blockDim.x + threadIdx.x;      // 4096 threads
  const float* r = cb + (size_t)t * DIM;
  float s = 0.f;
  for (int d = 0; d < DIM; d += 4) {
    float4 v = *(const float4*)(r + d);
    s += v.x * v.x + v.y * v.y + v.z * v.z + v.w * v.w;
  }
  wn[t] = s;
}

// ---------------- pairwise-distance sum per codebook ----------------
__global__ void pdist_k(const float* __restrict__ cb, float* __restrict__ pairs) {
  __shared__ float si[16 * LDS_PITCH];
  __shared__ float sj[16 * LDS_PITCH];
  __shared__ float red[256];
  int bi = blockIdx.x, bj = blockIdx.y, c = blockIdx.z;
  if (bj < bi) return;
  int tid = threadIdx.y * 16 + threadIdx.x;
  const float* base = cb + (size_t)c * K_CB * DIM;
  for (int k = tid; k < 16 * DIM; k += 256) {
    int r = k >> 8, d = k & 255;
    si[r * LDS_PITCH + d] = base[(size_t)(bi * 16 + r) * DIM + d];
    sj[r * LDS_PITCH + d] = base[(size_t)(bj * 16 + r) * DIM + d];
  }
  __syncthreads();
  int i = bi * 16 + threadIdx.y;
  int j = bj * 16 + threadIdx.x;
  float dsum = 0.f;
  if (i < j) {
    float acc = 0.f;
    for (int d = 0; d < DIM; ++d) {
      float t = si[threadIdx.y * LDS_PITCH + d] - sj[threadIdx.x * LDS_PITCH + d];
      acc += t * t;
    }
    dsum = sqrtf(acc);
  }
  red[tid] = dsum;
  __syncthreads();
  for (int s = 128; s > 0; s >>= 1) { if (tid < s) red[tid] += red[tid + s]; __syncthreads(); }
  if (tid == 0) atomicAdd(&pairs[c], red[0]);
}

// ---------------- fused VQ level: WMMA scores + argmin + gather + residual update ----------------
__global__ void vq_k(const float* __restrict__ rin, const float* __restrict__ cb,
                     const __bf16* __restrict__ bp, const float* __restrict__ wn,
                     float* __restrict__ rout, int* __restrict__ idx_ws,
                     unsigned* __restrict__ counts, float* __restrict__ sumsq, int c) {
  extern __shared__ float smem[];
  const int lane  = threadIdx.x & 31;
  const int wave  = threadIdx.x >> 5;
  const int m0    = blockIdx.x * 128 + wave * 16;     // 16 rows per wave
  const int wbase = wave * 16 * LDS_PITCH;

  // stage this wave's 16 fp32 rows into LDS (coalesced float4)
  for (int k = lane; k < 1024; k += 32) {
    int r = k >> 6, j = k & 63;
    float4 v = *(const float4*)(rin + (size_t)(m0 + r) * DIM + j * 4);
    *(float4*)&smem[wbase + r * LDS_PITCH + j * 4] = v;
  }
  __syncthreads();

  // build A fragments: 8 chunks of 16x32 bf16 (contiguous K per lane half, matching packed B)
  v16bf a[8];
  const int rl   = lane & 15;
  const int koff = (lane & 16) ? 16 : 0;
#pragma unroll
  for (int kc = 0; kc < 8; ++kc) {
#pragma unroll
    for (int h = 0; h < 16; ++h) {
      int K = kc * 32 + h + koff;
      a[kc][h] = (__bf16)smem[wbase + rl * LDS_PITCH + K];
    }
  }

  const float*  wnc = wn + c * K_CB;
  const __bf16* bpc = bp + (size_t)c * 64 * 8 * 512;

  float bestv[8]; int besti[8];
#pragma unroll
  for (int v = 0; v < 8; ++v) { bestv[v] = 3.4e38f; besti[v] = 0; }

  for (int nt = 0; nt < 64; ++nt) {
    const __bf16* bt = bpc + (size_t)nt * 8 * 512 + lane * 16;
    __builtin_prefetch(bt + 8 * 512, 0, 0);           // next tile -> global_prefetch
    v8f acc = {};
#pragma unroll
    for (int kc = 0; kc < 8; ++kc) {
      v16bf b = *(const v16bf*)(bt + kc * 512);
      acc = __builtin_amdgcn_wmma_f32_16x16x32_bf16(false, a[kc], false, b,
                                                    (short)0, acc, false, false);
    }
    int col = nt * 16 + rl;
    float w2 = wnc[col];
#pragma unroll
    for (int v = 0; v < 8; ++v) {
      float s = w2 - 2.f * acc[v];                    // ||w||^2 - 2 r.w  (argmin-equivalent)
      if (s < bestv[v]) { bestv[v] = s; besti[v] = col; }
    }
  }

  // min-reduce across the 16 lanes of each half-wave (column classes)
#pragma unroll
  for (int v = 0; v < 8; ++v) {
#pragma unroll
    for (int m = 8; m >= 1; m >>= 1) {
      float ov = __shfl_xor(bestv[v], m, 32);
      int   oi = __shfl_xor(besti[v], m, 32);
      if (ov < bestv[v] || (ov == bestv[v] && oi < besti[v])) { bestv[v] = ov; besti[v] = oi; }
    }
  }

  // lane 0 -> rows m0..m0+7 , lane 16 -> rows m0+8..m0+15
  if ((lane & 15) == 0) {
    int rb = m0 + ((lane >> 4) << 3);
#pragma unroll
    for (int v = 0; v < 8; ++v) {
      idx_ws[(size_t)c * N_VEC + rb + v] = besti[v];
      atomicAdd(&counts[c * K_CB + besti[v]], 1u);
    }
  }

  // broadcast all 16 row indices to every lane
  int rowIdx[16];
#pragma unroll
  for (int v = 0; v < 8; ++v) rowIdx[v]     = __shfl(besti[v], 0, 32);
#pragma unroll
  for (int v = 0; v < 8; ++v) rowIdx[v + 8] = __shfl(besti[v], 16, 32);

  // fused update: r_new = r_old - W[idx];  quant loss uses sum(r_new^2)
  const float* cbc = cb + (size_t)c * K_CB * DIM;
  float ss = 0.f;
#pragma unroll
  for (int r = 0; r < 16; ++r) {
    const float4* q4 = (const float4*)(cbc + (size_t)rowIdx[r] * DIM + lane * 8);
    const float4* s4 = (const float4*)&smem[wbase + r * LDS_PITCH + lane * 8];
    float4*       o4 = (float4*)(rout + (size_t)(m0 + r) * DIM + lane * 8);
#pragma unroll
    for (int p = 0; p < 2; ++p) {
      float4 q = q4[p], sv = s4[p], rn;
      rn.x = sv.x - q.x; rn.y = sv.y - q.y; rn.z = sv.z - q.z; rn.w = sv.w - q.w;
      ss += rn.x * rn.x + rn.y * rn.y + rn.z * rn.z + rn.w * rn.w;
      o4[p] = rn;
    }
  }
#pragma unroll
  for (int m = 16; m >= 1; m >>= 1) ss += __shfl_xor(ss, m, 32);
  if (lane == 0) atomicAdd(&sumsq[c], ss);
}

// ---------------- finalize: quantized = x - residual (in place over d_out) ----------------
__global__ void fin_k(const float* __restrict__ x, float* __restrict__ qout) {
  size_t i = ((size_t)blockIdx.x * blockDim.x + threadIdx.x) * 4;
  float4 xv = *(const float4*)(x + i);
  float4 rv = *(const float4*)(qout + i);
  float4 q;
  q.x = xv.x - rv.x; q.y = xv.y - rv.y; q.z = xv.z - rv.z; q.w = xv.w - rv.w;
  *(float4*)(qout + i) = q;
}

__global__ void idxout_k(const int* __restrict__ idx, float* __restrict__ out) {
  int t = blockIdx.x * blockDim.x + threadIdx.x;      // 262144 threads
  out[t] = (float)idx[t];
}

// ---------------- scalars: quant / util / compact ----------------
__global__ void scalars_k(const unsigned* __restrict__ counts, const float* __restrict__ sumsq,
                          const float* __restrict__ pairs, float* __restrict__ out) {
  __shared__ float red[256];
  int t = threadIdx.x;
  float u = 0.f;
  for (int k = t; k < LEVELS * K_CB; k += 256) {
    float d = (float)counts[k] - (float)(N_VEC / K_CB);
    u += fabsf(d);
  }
  red[t] = u;
  __syncthreads();
  for (int s = 128; s > 0; s >>= 1) { if (t < s) red[t] += red[t + s]; __syncthreads(); }
  if (t == 0) {
    float qs = 0.f, cs = 0.f;
    for (int c = 0; c < LEVELS; ++c) { qs += sumsq[c]; cs += pairs[c]; }
    const float npairs = (float)(K_CB * (K_CB - 1) / 2);
    out[0] = 1.25f * qs / ((float)N_VEC * (float)DIM);  // quant
    out[1] = red[0] / (float)K_CB;                      // util
    out[2] = 2.f * cs / npairs;                         // compact
  }
}

extern "C" void kernel_launch(void* const* d_in, const int* in_sizes, int n_in,
                              void* d_out, int out_size, void* d_ws, size_t ws_size,
                              hipStream_t stream) {
  (void)in_sizes; (void)n_in; (void)out_size; (void)ws_size;
  const float* x  = (const float*)d_in[0];
  const float* cb = (const float*)d_in[1];
  float* out = (float*)d_out;

  char* ws = (char*)d_ws;
  __bf16*   bpack  = (__bf16*)(ws + WS_BPACK);
  float*    wnorm  = (float*)(ws + WS_WNORM);
  unsigned* counts = (unsigned*)(ws + WS_COUNTS);
  int*      idxw   = (int*)(ws + WS_IDX);
  float*    sumsq  = (float*)(ws + WS_SUMSQ);
  float*    pairs  = (float*)(ws + WS_PAIRS);

  float* resid = out;   // quantized region doubles as residual scratch; fin_k converts in place

  init_k<<<1, 256, 0, stream>>>(counts, sumsq, pairs);
  pack_k<<<256, 256, 0, stream>>>(cb, bpack);
  wnorm_k<<<16, 256, 0, stream>>>(cb, wnorm);
  pdist_k<<<dim3(64, 64, LEVELS), dim3(16, 16, 1), 0, stream>>>(cb, pairs);

  size_t lds = (size_t)8 * 16 * LDS_PITCH * sizeof(float);   // 130 KB / block (320 KB WGP LDS)
  for (int c = 0; c < LEVELS; ++c) {
    vq_k<<<512, 256, lds, stream>>>(c == 0 ? x : resid, cb, bpack, wnorm,
                                    resid, idxw, counts, sumsq, c);
  }

  fin_k<<<16384, 256, 0, stream>>>(x, out);                 // quantized = x - residual
  idxout_k<<<1024, 256, 0, stream>>>(idxw, out + OUT_IDX);
  scalars_k<<<1, 256, 0, stream>>>(counts, sumsq, pairs, out + OUT_SCAL);
}
